// GPT_38328288150355
// MI455X (gfx1250) — matrix operand order repaired and checked by reading, not compiled, not run on hardware
//
#include <hip/hip_runtime.h>
#include <cstdint>
#include <cstddef>

// GPT-2 small geometry (fixed by the reference)
#define BB   2
#define TT   1024
#define LL   12
#define HH   12
#define CC   768
#define VV   50304
#define HDIM 64
#define MTOK (BB*TT)   // 2048 token rows

typedef __attribute__((ext_vector_type(16))) _Float16 v16h;
typedef __attribute__((ext_vector_type(8)))  _Float16 v8h;
typedef __attribute__((ext_vector_type(8)))  float    v8f;

// ---------------------------------------------------------------------------
// Embedding: x[b,t,:] = wte[idx[b,t],:] + wpe[t,:]
// ---------------------------------------------------------------------------
__global__ void embed_kernel(const int* __restrict__ idx,
                             const float* __restrict__ wte,
                             const float* __restrict__ wpe,
                             float* __restrict__ x)
{
    int i   = blockIdx.x * blockDim.x + threadIdx.x;  // < MTOK*CC exactly
    int c   = i % CC;
    int tok = i / CC;            // 0..MTOK-1
    int t   = tok % TT;
    int id  = idx[tok];
    x[i] = wte[(size_t)id * CC + c] + wpe[(size_t)t * CC + c];
}

// ---------------------------------------------------------------------------
// LayerNorm over C=768, one row per 256-thread block
// ---------------------------------------------------------------------------
__global__ void layernorm_kernel(const float* __restrict__ x,
                                 const float* __restrict__ w,
                                 const float* __restrict__ b,
                                 float* __restrict__ y)
{
    __shared__ float rs[256];
    __shared__ float rs2[256];
    const int row = blockIdx.x;
    const int tid = threadIdx.x;
    const float* xr = x + (size_t)row * CC;
    float s = 0.f, s2 = 0.f;
    for (int i = tid; i < CC; i += 256) { float v = xr[i]; s += v; s2 += v * v; }
    rs[tid] = s; rs2[tid] = s2;
    __syncthreads();
    for (int o = 128; o > 0; o >>= 1) {
        if (tid < o) { rs[tid] += rs[tid + o]; rs2[tid] += rs2[tid + o]; }
        __syncthreads();
    }
    const float mean = rs[0] * (1.f / CC);
    const float var  = rs2[0] * (1.f / CC) - mean * mean;
    const float rstd = rsqrtf(var + 1e-5f);
    float* yr = y + (size_t)row * CC;
    for (int i = tid; i < CC; i += 256)
        yr[i] = (xr[i] - mean) * rstd * w[i] + b[i];
}

// ---------------------------------------------------------------------------
// WMMA GEMM: Y = X(MxK) @ W(NxK)^T  [+ bias] [+ res] [GELU]
// Block: 256 threads = 8 waves, 128x128 output tile.
// Wave (wm 0..3 -> 32 M-rows, wn 0..1 -> 64 N-cols): 8 WMMAs per 32-K step,
// each B fragment reused for two A fragments (halves weight-stream bytes/FLOP).
// Grid: M in blockIdx.x (fastest) so concurrent blocks share the same weight
// slice -> weights stream from HBM once, re-reads hit the 192 MB L2.
// A tile (128x32) staged via LDS as f16 (coalesced f32 loads, shared by waves).
// B fragment: lane n=lane&15 holds 16 consecutive K (k = 16*(lane>=16)+half),
// a contiguous 16-float run of the row-major (N,K) weight.
// ---------------------------------------------------------------------------
#define EPI_BIAS       0
#define EPI_BIAS_RES   1
#define EPI_BIAS_GELU  2
#define EPI_NONE       3

__device__ inline void cvt4(v16h& v, int base, float4 f) {
    v[base + 0] = (_Float16)f.x; v[base + 1] = (_Float16)f.y;
    v[base + 2] = (_Float16)f.z; v[base + 3] = (_Float16)f.w;
}

template<int EPI>
__global__ __launch_bounds__(256)
void wmma_gemm(const float* __restrict__ X,     // M x K
               const float* __restrict__ W,     // N x K (row-major)
               const float* __restrict__ bias,  // N (unused for EPI_NONE)
               const float* __restrict__ res,   // M x N (EPI_BIAS_RES only)
               float* __restrict__ Y,           // M x N
               int M, int N, int K)
{
    __shared__ _Float16 lds_a[128 * 32];        // 128 rows x 32 k, f16 (8 KB)

    const int tid  = threadIdx.x;
    const int lane = tid & 31;
    const int wave = tid >> 5;                  // 0..7
    const int wm   = wave & 3;                  // 4 waves along M (32 rows each)
    const int wn   = wave >> 2;                 // 2 waves along N (64 cols each)
    const int block_m = blockIdx.x * 128;       // M fastest-varying for L2 reuse
    const int tile_m  = block_m + wm * 32;
    const int tile_n  = blockIdx.y * 128 + wn * 64;

    // A fragment mapping (16-bit A 16x32, this code's contiguous k order, same as B):
    //   lanes 0-15 : M=lane,    halves 0-15 -> K 0..15
    //   lanes 16-31: M=lane-16, halves 0-15 -> K 16..31
    const int arow_l = wm * 32 + (lane & 15);   // row within LDS tile (mi adds 16)
    const int akoff  = (lane >> 4) * 16;        // 0 or 16

    // B fragment mapping (16-bit B KxN): lane = n + 16*(k>=16), half = k&15
    const int bn    = lane & 15;
    const int bkoff = (lane >> 4) * 16;         // 0 or 16

    const float* bptr0 = W + (size_t)(tile_n +  0 + bn) * K + bkoff;
    const float* bptr1 = W + (size_t)(tile_n + 16 + bn) * K + bkoff;
    const float* bptr2 = W + (size_t)(tile_n + 32 + bn) * K + bkoff;
    const float* bptr3 = W + (size_t)(tile_n + 48 + bn) * K + bkoff;

    v8f acc[2][4] = {};

    const int r0 = tid >> 3;                    // 0..31 (staging row)
    const int ch = tid & 7;                     // float4 chunk within 32 k

    for (int k0 = 0; k0 < K; k0 += 32) {
        // ---- stage A tile (128 x 32) to LDS as f16, coalesced ----
        #pragma unroll
        for (int rr = 0; rr < 128; rr += 32) {
            const int r = r0 + rr;
            float4 f = *reinterpret_cast<const float4*>(
                X + (size_t)(block_m + r) * K + k0 + ch * 4);
            _Float16* d = &lds_a[r * 32 + ch * 4];
            d[0] = (_Float16)f.x; d[1] = (_Float16)f.y;
            d[2] = (_Float16)f.z; d[3] = (_Float16)f.w;
        }
        // prefetch next K-step of the weight stream (global_prefetch_b8)
        if (k0 + 32 < K) {
            __builtin_prefetch(bptr0 + k0 + 32, 0, 1);
            __builtin_prefetch(bptr1 + k0 + 32, 0, 1);
            __builtin_prefetch(bptr2 + k0 + 32, 0, 1);
            __builtin_prefetch(bptr3 + k0 + 32, 0, 1);
        }
        __syncthreads();

        // ---- read two A fragments (rows +0, +16) from LDS ----
        v16h afrag[2];
        #pragma unroll
        for (int mi = 0; mi < 2; ++mi) {
            const int ar = arow_l + mi * 16;
            v8h alo = *reinterpret_cast<const v8h*>(&lds_a[ar * 32 + akoff]);
            v8h ahi = *reinterpret_cast<const v8h*>(&lds_a[ar * 32 + akoff + 8]);
            #pragma unroll
            for (int i = 0; i < 8; ++i) {
                afrag[mi][i] = alo[i]; afrag[mi][i + 8] = ahi[i];
            }
        }

        // ---- 4 B fragments, each driving 2 WMMAs ----
        #pragma unroll
        for (int s = 0; s < 4; ++s) {
            const float* bp = (s == 0) ? bptr0 : (s == 1) ? bptr1
                            : (s == 2) ? bptr2 : bptr3;
            const float4* bv = reinterpret_cast<const float4*>(bp + k0);
            v16h bfrag;
            cvt4(bfrag,  0, bv[0]);
            cvt4(bfrag,  4, bv[1]);
            cvt4(bfrag,  8, bv[2]);
            cvt4(bfrag, 12, bv[3]);
            #pragma unroll
            for (int mi = 0; mi < 2; ++mi) {
                acc[mi][s] = __builtin_amdgcn_wmma_f32_16x16x32_f16(
                    false, afrag[mi], false, bfrag, (short)0, acc[mi][s],
                    false, false);
            }
        }
        __syncthreads();
    }

    // ---- epilogue: C/D layout VGPR r, lane -> (M = r + 8*(lane>=16), N = lane&15)
    #pragma unroll
    for (int mi = 0; mi < 2; ++mi) {
        const int row0 = tile_m + mi * 16 + (lane >> 4) * 8;
        #pragma unroll
        for (int s = 0; s < 4; ++s) {
            v8f a = acc[mi][s];
            const int col = tile_n + 16 * s + (lane & 15);
            const float bvadd = (EPI == EPI_NONE) ? 0.f : bias[col];
            #pragma unroll
            for (int r = 0; r < 8; ++r) {
                const size_t o = (size_t)(row0 + r) * N + col;
                float v = a[r] + bvadd;
                if (EPI == EPI_BIAS_RES)  v += res[o];
                if (EPI == EPI_BIAS_GELU) v = 0.5f * v * (1.f + erff(v * 0.70710678118654752f));
                Y[o] = v;
            }
        }
    }
}

// ---------------------------------------------------------------------------
// Causal attention (VALU; ~39 GFLOP total, tiny vs GEMMs).
// One 128-thread block per (b, h, q). qkv layout: (B,T,3C), head offset h*64.
// ---------------------------------------------------------------------------
__global__ void attention_kernel(const float* __restrict__ qkv,
                                 float* __restrict__ y)
{
    __shared__ float qs[HDIM];
    __shared__ float sc[TT];
    __shared__ float rbuf[128];
    const int tid = threadIdx.x;
    const int bhq = blockIdx.x;
    const int q = bhq % TT;
    const int h = (bhq / TT) % HH;
    const int b = bhq / (TT * HH);

    const float* base = qkv + (size_t)b * TT * (3 * CC);
    const float* qrow = base + (size_t)q * (3 * CC) + h * HDIM;
    if (tid < HDIM) qs[tid] = qrow[tid];
    __syncthreads();

    const float scale = rsqrtf((float)HDIM);
    float lmax = -3.4e38f;
    for (int j = tid; j <= q; j += 128) {
        const float* kr = base + (size_t)j * (3 * CC) + CC + h * HDIM;
        float d = 0.f;
        #pragma unroll
        for (int e = 0; e < HDIM; ++e) d += qs[e] * kr[e];
        d *= scale;
        sc[j] = d;
        lmax = fmaxf(lmax, d);
    }
    rbuf[tid] = lmax;
    __syncthreads();
    for (int o = 64; o > 0; o >>= 1) {
        if (tid < o) rbuf[tid] = fmaxf(rbuf[tid], rbuf[tid + o]);
        __syncthreads();
    }
    const float mx = rbuf[0];
    __syncthreads();

    float lsum = 0.f;
    for (int j = tid; j <= q; j += 128) {
        float p = __expf(sc[j] - mx);
        sc[j] = p;
        lsum += p;
    }
    rbuf[tid] = lsum;
    __syncthreads();
    for (int o = 64; o > 0; o >>= 1) {
        if (tid < o) rbuf[tid] += rbuf[tid + o];
        __syncthreads();
    }
    const float inv = 1.f / rbuf[0];
    __syncthreads();

    if (tid < HDIM) {
        float a = 0.f;
        for (int j = 0; j <= q; ++j) {
            const float* vr = base + (size_t)j * (3 * CC) + 2 * CC + h * HDIM;
            a += sc[j] * vr[tid];   // coalesced across the 64 lanes
        }
        y[((size_t)(b * TT + q)) * CC + h * HDIM + tid] = a * inv;
    }
}

// ---------------------------------------------------------------------------
// Host orchestration (graph-capture safe: launches only)
// ---------------------------------------------------------------------------
extern "C" void kernel_launch(void* const* d_in, const int* in_sizes, int n_in,
                              void* d_out, int out_size, void* d_ws, size_t ws_size,
                              hipStream_t stream)
{
    (void)in_sizes; (void)n_in; (void)out_size; (void)ws_size;

    const int*   idx    = (const int*)  d_in[0];
    const float* wte    = (const float*)d_in[1];
    const float* wpe    = (const float*)d_in[2];
    const float* ln1_w  = (const float*)d_in[3];
    const float* ln1_b  = (const float*)d_in[4];
    const float* attn_w = (const float*)d_in[5];
    const float* attn_b = (const float*)d_in[6];
    const float* apw    = (const float*)d_in[7];
    const float* apb    = (const float*)d_in[8];
    const float* ln2_w  = (const float*)d_in[9];
    const float* ln2_b  = (const float*)d_in[10];
    const float* fc_w   = (const float*)d_in[11];
    const float* fc_b   = (const float*)d_in[12];
    const float* mpw    = (const float*)d_in[13];
    const float* mpb    = (const float*)d_in[14];
    const float* lnf_w  = (const float*)d_in[15];
    const float* lnf_b  = (const float*)d_in[16];

    const int M = MTOK;                          // 2048
    float* ws   = (float*)d_ws;
    float* x    = ws;                            // M*C
    float* hbuf = x    + (size_t)M * CC;         // M*C
    float* qkv  = hbuf + (size_t)M * CC;         // M*3C
    float* yat  = qkv  + (size_t)M * 3 * CC;     // M*C
    float* mid  = yat  + (size_t)M * CC;         // M*4C
    float* logits = (float*)d_out;               // M*V

    const dim3 blk(256);

    embed_kernel<<<(M * CC) / 256, blk, 0, stream>>>(idx, wte, wpe, x);

    for (int l = 0; l < LL; ++l) {
        layernorm_kernel<<<M, blk, 0, stream>>>(
            x, ln1_w + (size_t)l * CC, ln1_b + (size_t)l * CC, hbuf);

        wmma_gemm<EPI_BIAS><<<dim3(M / 128, (3 * CC) / 128), blk, 0, stream>>>(
            hbuf, attn_w + (size_t)l * 3 * CC * CC, attn_b + (size_t)l * 3 * CC,
            nullptr, qkv, M, 3 * CC, CC);

        attention_kernel<<<BB * HH * TT, 128, 0, stream>>>(qkv, yat);

        wmma_gemm<EPI_BIAS_RES><<<dim3(M / 128, CC / 128), blk, 0, stream>>>(
            yat, apw + (size_t)l * CC * CC, apb + (size_t)l * CC,
            x, x, M, CC, CC);

        layernorm_kernel<<<M, blk, 0, stream>>>(
            x, ln2_w + (size_t)l * CC, ln2_b + (size_t)l * CC, hbuf);

        wmma_gemm<EPI_BIAS_GELU><<<dim3(M / 128, (4 * CC) / 128), blk, 0, stream>>>(
            hbuf, fc_w + (size_t)l * 4 * CC * CC, fc_b + (size_t)l * 4 * CC,
            nullptr, mid, M, 4 * CC, CC);

        wmma_gemm<EPI_BIAS_RES><<<dim3(M / 128, CC / 128), blk, 0, stream>>>(
            mid, mpw + (size_t)l * CC * 4 * CC, mpb + (size_t)l * CC,
            x, x, M, CC, 4 * CC);
    }

    layernorm_kernel<<<M, blk, 0, stream>>>(x, lnf_w, lnf_b, hbuf);

    // tied lm_head: logits = hbuf @ wte^T   (N = 50304 = 393 * 128)
    wmma_gemm<EPI_NONE><<<dim3(M / 128, VV / 128), blk, 0, stream>>>(
        hbuf, wte, nullptr, nullptr, logits, M, VV, CC);
}
